// MultiScaleDynamicWindow3D_85478439125358
// MI455X (gfx1250) — matrix operand order repaired and checked by reading, not compiled, not run on hardware
//
#include <hip/hip_runtime.h>
#include <hip/hip_fp16.h>
#include <math.h>

typedef __attribute__((ext_vector_type(16))) _Float16 v16h;
typedef __attribute__((ext_vector_type(8)))  float    v8f;

#define SPATIAL (64*64*64)       // 262144 = 2^18
#define NBATCH  2
#define NCH     48
#define NVOX    (NBATCH*SPATIAL) // 524288
#define NSCALE  3

// ---------------------------------------------------------------------------
// Pass 1: y[s][v] = sum_c fuse_w[s*48+c] * x[b,c,sp] via v_wmma_f32_16x16x32_f16.
// Each wave iteration: 32 consecutive voxels = two 16-column N-tiles,
// K=48 split as (ch 0..31) + (ch 32..47, upper half-wave zero-masked).
// All loads are unconditional (flat control flow) from a readfirstlane'd
// scalar base -> saddr-form coalesced b32 loads, full clauses in flight.
// ---------------------------------------------------------------------------
__global__ __launch_bounds__(256)
void msdw_reduce_wmma(const float* __restrict__ x,
                      const float* __restrict__ fw,
                      float* __restrict__ y) {
  const int lane = threadIdx.x & 31;
  const int half = lane >> 4;            // 0: lanes 0-15, 1: lanes 16-31
  const int m    = lane & 15;            // A row / B column (voxel-in-tile)
  const float bmask = (half == 0) ? 1.0f : 0.0f;  // zero K=48..63 padding

  // ---- A fragments (weights): contiguous K = 16*half + e per half-wave ----
  v16h a0, a1;
#pragma unroll
  for (int e = 0; e < 16; ++e) {
    const int K  = e + half * 16;  // {0..15}(+16 hi)
    const float w0 = (m < NSCALE) ? fw[m * NCH + K] : 0.0f;
    const int c1 = 32 + K;
    const float w1 = (m < NSCALE && c1 < NCH) ? fw[m * NCH + c1] : 0.0f;
    a0[e] = (_Float16)w0;
    a1[e] = (_Float16)w1;
  }

  const int nwaves = (gridDim.x * blockDim.x) >> 5;
  const int wid    = (blockIdx.x * blockDim.x + threadIdx.x) >> 5;
  const int npairs = NVOX / 32;          // 16384 double-tiles

  for (int t = wid; t < npairs; t += nwaves) {
    const int v0 = t * 32;               // 32 voxels, never crosses batch
    const int b  = v0 >> 18;             // / SPATIAL
    const int s0 = v0 & (SPATIAL - 1);
    unsigned base = ((unsigned)b * (unsigned)(NCH * SPATIAL) + (unsigned)s0) * 4u;
    base = (unsigned)__builtin_amdgcn_readfirstlane((int)base);  // scalarize
    const float* px = (const float*)((const char*)x + base);

    // ---- B fragments: column = m, K contiguous per half-wave ----
    v16h b0A, b1A, b0B, b1B;
#pragma unroll
    for (int e = 0; e < 16; ++e) {
      const size_t off0 = (size_t)(e + half * 16) * SPATIAL + m;  // ch 0..31
      const size_t off1 = (size_t)(32 + e) * SPATIAL + m;         // ch 32..47
      // tile A: voxels v0+0..15 ; tile B: voxels v0+16..31
      b0A[e] = (_Float16)px[off0];
      b0B[e] = (_Float16)px[off0 + 16];
      // both half-waves read the SAME 64B line (no divergence, no extra
      // traffic); upper half masked to zero (K=48..63 padding).
      b1A[e] = (_Float16)(px[off1]      * bmask);
      b1B[e] = (_Float16)(px[off1 + 16] * bmask);
    }

    v8f accA = {}, accB = {};
    accA = __builtin_amdgcn_wmma_f32_16x16x32_f16(false, a0, false, b0A,
                                                  (short)0, accA, false, false);
    accA = __builtin_amdgcn_wmma_f32_16x16x32_f16(false, a1, false, b1A,
                                                  (short)0, accA, false, false);
    accB = __builtin_amdgcn_wmma_f32_16x16x32_f16(false, a0, false, b0B,
                                                  (short)0, accB, false, false);
    accB = __builtin_amdgcn_wmma_f32_16x16x32_f16(false, a1, false, b1B,
                                                  (short)0, accB, false, false);

    // D rows 0..2 live in acc[0..2] of lanes 0-15 (column N = m)
    if (half == 0) {
#pragma unroll
      for (int s = 0; s < NSCALE; ++s) {
        y[(size_t)s * NVOX + v0 + m]      = accA[s];
        y[(size_t)s * NVOX + v0 + 16 + m] = accB[s];
      }
    }
  }
}

// ---------------------------------------------------------------------------
// Passes 2-3: 1D box sums (zero-padded) along d (shift=12) and h (shift=6),
// per-scale radius r = s+1 (k = 3,5,7). Layout: [scale][b][d][h][w].
// ---------------------------------------------------------------------------
__global__ void msdw_box(const float* __restrict__ in, float* __restrict__ out,
                         int shift) {
  const int i = blockIdx.x * blockDim.x + threadIdx.x;
  if (i >= NSCALE * NVOX) return;
  const int s   = i >> 19;               // / NVOX (2^19)
  const int v   = i & (NVOX - 1);
  const int r   = s + 1;
  const int sp  = v & (SPATIAL - 1);
  const int pos = (sp >> shift) & 63;
  const int lo  = (pos - r < 0  ? 0  : pos - r) - pos;
  const int hi  = (pos + r > 63 ? 63 : pos + r) - pos;
  const int stride = 1 << shift;
  const float* p = in + i;
  float sum = 0.0f;
  for (int dz = lo; dz <= hi; ++dz) sum += p[dz * stride];
  out[i] = sum;
}

// ---------------------------------------------------------------------------
// Pass 4: box sum along w, combine scales with 1/k^3, bias, sigmoid.
// ---------------------------------------------------------------------------
__global__ void msdw_final(const float* __restrict__ in,
                           const float* __restrict__ bias,
                           float* __restrict__ out) {
  const int v = blockIdx.x * blockDim.x + threadIdx.x;
  if (v >= NVOX) return;
  const int w = v & 63;
  const float invk3[3] = {1.0f/27.0f, 1.0f/125.0f, 1.0f/343.0f};
  float t = bias[0];
#pragma unroll
  for (int s = 0; s < NSCALE; ++s) {
    const int r  = s + 1;
    const int lo = (w - r < 0  ? 0  : w - r) - w;
    const int hi = (w + r > 63 ? 63 : w + r) - w;
    const float* p = in + (size_t)s * NVOX + v;
    float sum = 0.0f;
    for (int dz = lo; dz <= hi; ++dz) sum += p[dz];
    t += sum * invk3[s];
  }
  out[v] = 1.0f / (1.0f + __expf(-t));
}

// ---------------------------------------------------------------------------
extern "C" void kernel_launch(void* const* d_in, const int* in_sizes, int n_in,
                              void* d_out, int out_size, void* d_ws, size_t ws_size,
                              hipStream_t stream) {
  const float* x  = (const float*)d_in[0];   // (2,48,64,64,64) f32
  const float* fw = (const float*)d_in[1];   // (1,144) f32
  const float* fb = (const float*)d_in[2];   // (1,)   f32
  float* out  = (float*)d_out;               // (2,1,64,64,64) f32

  float* bufA = (float*)d_ws;                // 3*NVOX floats
  float* bufB = bufA + (size_t)NSCALE * NVOX;

  // Pass 1: WMMA channel reduction (8192 waves, 2 double-tiles each)
  msdw_reduce_wmma<<<1024, 256, 0, stream>>>(x, fw, bufA);

  const int n3  = NSCALE * NVOX;
  const int blk = 256;
  msdw_box<<<(n3 + blk - 1) / blk, blk, 0, stream>>>(bufA, bufB, 12); // along d
  msdw_box<<<(n3 + blk - 1) / blk, blk, 0, stream>>>(bufB, bufA, 6);  // along h
  msdw_final<<<(NVOX + blk - 1) / blk, blk, 0, stream>>>(bufA, fb, out); // along w
}
